// LSHBlockAttention_47150150975523
// MI455X (gfx1250) — matrix operand order repaired and checked by reading, not compiled, not run on hardware
//
#include <hip/hip_runtime.h>
#include <hip/hip_bf16.h>
#include <math.h>

typedef __attribute__((ext_vector_type(16))) _Float16 v16h;
typedef __attribute__((ext_vector_type(8)))  float    v8f;

#define B_   4
#define N_   4096
#define C_   1024
#define H_   16
#define DH_  64
#define NB_  64
#define BN_  (B_ * N_)
// finfo(float32).min / 4
#define LNEG (-3.4028234663852886e38f / 4.0f)

// Padded LDS row stride (halves) for WMMA tiles: 72 halves = 144 B = 36 dwords.
// 36*i mod 64 distinct for i in [0,16) -> conflict-free b128 fragment loads,
// and 144 % 16 == 0 keeps every 16B chunk aligned for async/ds b128.
#define PAD_ 72

// ---------------------------------------------------------------------------
// CDNA5 async global->LDS copy (ASYNCcnt) + wait, via inline asm (portable
// across ROCm-7.2 / amdgpu-toolchain builtin arity differences).
// Per lane: LDS[ldsoff] = MEM[gaddr], 16 bytes.
// ---------------------------------------------------------------------------
__device__ inline void async_copy_b128(unsigned ldsoff, unsigned long long gaddr) {
  asm volatile("global_load_async_to_lds_b128 %0, %1, off"
               :: "v"(ldsoff), "v"(gaddr) : "memory");
}
__device__ inline void wait_async0() {
  asm volatile("s_wait_asynccnt 0x0" ::: "memory");
}
template <typename T>
__device__ inline unsigned lds_off(T* p) {          // flat addr low 32b == LDS offset
  return (unsigned)(uintptr_t)p;
}

// ---------------------------------------------------------------------------
// WMMA fragment loaders (CDNA5 ISA 7.12.2, wave32)
// A (16x32 f16): lane L -> row M=L%16; elems 0..7 = K[(L/16)*8..], 8..15 = K[16+(L/16)*8..]
// B (32x16 f16): lane L -> col N=L%16; elems 0..7 = K[(L/16)*8..], 8..15 = K[16+(L/16)*8..]
// C/D (16x16 f32): elem e -> row (L<16 ? e : e+8), col L%16
// ---------------------------------------------------------------------------
__device__ inline v16h load_a_frag(const _Float16* __restrict__ rowk0, int lh) {
  v16h r;
  ((uint4*)&r)[0] = *(const uint4*)(rowk0 + lh * 8);
  ((uint4*)&r)[1] = *(const uint4*)(rowk0 + 16 + lh * 8);
  return r;
}
__device__ inline v16h load_b_frag(const _Float16* __restrict__ colk0, int lh) {
  v16h r;
  ((uint4*)&r)[0] = *(const uint4*)(colk0 + lh * 8);
  ((uint4*)&r)[1] = *(const uint4*)(colk0 + 16 + lh * 8);
  return r;
}

// ---------------------------------------------------------------------------
__global__ void cast_x_kernel(const float* __restrict__ in, _Float16* __restrict__ out, int n) {
  int i = blockIdx.x * blockDim.x + threadIdx.x;
  if (i < n) out[i] = (_Float16)in[i];
}

// Wt[n][k] = (f16) W[k][n]  so the GEMM B-operand is K-contiguous
__global__ void transpose_cast_kernel(const float* __restrict__ W, _Float16* __restrict__ Wt) {
  __shared__ float tile[32][33];
  int tx = threadIdx.x, ty = threadIdx.y;
  int bx = blockIdx.x * 32, by = blockIdx.y * 32;
#pragma unroll
  for (int i = 0; i < 4; ++i)
    tile[ty + i * 8][tx] = W[(size_t)(by + ty + i * 8) * C_ + bx + tx];
  __syncthreads();
#pragma unroll
  for (int i = 0; i < 4; ++i)
    Wt[(size_t)(bx + ty + i * 8) * C_ + by + tx] = (_Float16)tile[tx][ty + i * 8];
}

// ---------------------------------------------------------------------------
// LayerNorm statistics (fp32, exact argmax-critical path)
// ---------------------------------------------------------------------------
__global__ void ln_stats_kernel(const float* __restrict__ x,
                                float* __restrict__ mu, float* __restrict__ rs) {
  __shared__ float red[256];
  const int token = blockIdx.x;
  const int t = threadIdx.x;
  const float* xr = x + (size_t)token * C_;
  float s = 0.f;
  for (int i = t; i < C_; i += 256) s += xr[i];
  red[t] = s; __syncthreads();
  for (int w = 128; w > 0; w >>= 1) { if (t < w) red[t] += red[t + w]; __syncthreads(); }
  float m = red[0] * (1.0f / C_);
  __syncthreads();
  float v = 0.f;
  for (int i = t; i < C_; i += 256) { float d = xr[i] - m; v += d * d; }
  red[t] = v; __syncthreads();
  for (int w = 128; w > 0; w >>= 1) { if (t < w) red[t] += red[t + w]; __syncthreads(); }
  if (t == 0) {
    mu[token] = m;
    rs[token] = rsqrtf(red[0] * (1.0f / C_) + 1e-5f);
  }
}

// ---------------------------------------------------------------------------
// bucket_id[token] = first-max argmax_b  sum_c xn[c]*hp[c][b]  (fp32)
// ---------------------------------------------------------------------------
__global__ void lsh_bucket_kernel(const float* __restrict__ x, const float* __restrict__ hp,
                                  const float* __restrict__ mu, const float* __restrict__ rs,
                                  int* __restrict__ bid) {
  __shared__ float vals[NB_];
  const int token = blockIdx.x;
  const int t = threadIdx.x;
  const float* xr = x + (size_t)token * C_;
  const float m = mu[token], r = rs[token];
  float acc = 0.f;
  for (int c = 0; c < C_; ++c) acc += (xr[c] - m) * r * hp[(size_t)c * NB_ + t];
  vals[t] = acc;
  __syncthreads();
  if (t == 0) {
    float best = vals[0]; int bi = 0;
    for (int i = 1; i < NB_; ++i) { if (vals[i] > best) { best = vals[i]; bi = i; } }
    bid[token] = bi;
  }
}

// ---------------------------------------------------------------------------
// Stable counting sort per batch == argsort(bucket*(N+1)+pos)
// ---------------------------------------------------------------------------
__global__ void bucket_sort_kernel(const int* __restrict__ bid, int* __restrict__ perm) {
  __shared__ int cnt[NB_];
  __shared__ int off[NB_];
  const int b = blockIdx.x;
  const int t = threadIdx.x;
  cnt[t] = 0;
  __syncthreads();
  const int* bb = bid + (size_t)b * N_;
  for (int i = t; i < N_; i += NB_) atomicAdd(&cnt[bb[i]], 1);
  __syncthreads();
  if (t == 0) { int s = 0; for (int i = 0; i < NB_; ++i) { off[i] = s; s += cnt[i]; } }
  __syncthreads();
  int r = off[t];
  int* pb = perm + (size_t)b * N_;
  for (int i = 0; i < N_; ++i) if (bb[i] == t) pb[r++] = i;
}

// ---------------------------------------------------------------------------
// GEMM: out[M,N] = A[M,K](f16) * Bt[N,K]^T(f16) + bias[N]
// Block = 8 waves owning a 128x64 tile. Double-buffered async global->LDS
// copies (ASYNCcnt) overlap the WMMA math; LDS rows padded to 72 halves.
// Per K-step(64): each wave issues 6 async b128 copies (A:4, B:2, 512B each).
// ---------------------------------------------------------------------------
template <bool HALF_OUT>
__global__ __launch_bounds__(256) void gemm_wmma_kernel(
    const _Float16* __restrict__ A, const _Float16* __restrict__ Bt,
    const float* __restrict__ bias, void* __restrict__ outp, int M, int N, int K) {
  __shared__ __align__(16) _Float16 As[2][128 * PAD_];
  __shared__ __align__(16) _Float16 Bs[2][64 * PAD_];
  const int tid  = threadIdx.x;
  const int lane = tid & 31;
  const int wave = tid >> 5;
  const int lm = lane & 15, lh = lane >> 4;
  const int row0 = blockIdx.y * 128;
  const int col0 = blockIdx.x * 64;
  // async-copy lane mapping: lane covers (subrow, 16B chunk of a 128B row)
  const int subrow = lane >> 3;
  const int chunk  = lane & 7;

  v8f acc[4] = {};

  auto issue = [&](int buf, int k0) {
#pragma unroll
    for (int i = 0; i < 4; ++i) {          // A: rows wave*16 .. +15
      int r = wave * 16 + i * 4 + subrow;
      async_copy_b128(lds_off(&As[buf][r * PAD_ + chunk * 8]),
                      (unsigned long long)(uintptr_t)(A + (size_t)(row0 + r) * K + k0 + chunk * 8));
    }
#pragma unroll
    for (int i = 0; i < 2; ++i) {          // B: rows wave*8 .. +7
      int r = wave * 8 + i * 4 + subrow;
      async_copy_b128(lds_off(&Bs[buf][r * PAD_ + chunk * 8]),
                      (unsigned long long)(uintptr_t)(Bt + (size_t)(col0 + r) * K + k0 + chunk * 8));
    }
  };

  issue(0, 0);
  wait_async0();
  __syncthreads();

  for (int k0 = 0; k0 < K; k0 += 64) {
    const int buf = (k0 >> 6) & 1;
    if (k0 + 64 < K) issue(buf ^ 1, k0 + 64);
#pragma unroll
    for (int kk = 0; kk < 64; kk += 32) {
      v16h a = load_a_frag(&As[buf][(wave * 16 + lm) * PAD_ + kk], lh);
#pragma unroll
      for (int t = 0; t < 4; ++t) {
        v16h bf = load_b_frag(&Bs[buf][(t * 16 + lm) * PAD_ + kk], lh);
        acc[t] = __builtin_amdgcn_wmma_f32_16x16x32_f16(false, a, false, bf,
                                                        (short)0, acc[t], false, false);
      }
    }
    wait_async0();       // our next-buffer copies have landed
    __syncthreads();     // everyone done reading buf & writing buf^1
  }

#pragma unroll
  for (int t = 0; t < 4; ++t) {
    int col = col0 + t * 16 + lm;
    float bv = bias[col];
#pragma unroll
    for (int e = 0; e < 8; ++e) {
      int row = row0 + wave * 16 + (lh ? e + 8 : e);
      float v = acc[t][e] + bv;
      if (HALF_OUT) ((_Float16*)outp)[(size_t)row * N + col] = (_Float16)v;
      else          ((float*)outp)[(size_t)row * N + col] = v;
    }
  }
}

// ---------------------------------------------------------------------------
// Block-local LSH attention. One workgroup (4 waves) per (block, head, batch).
// Q/K gathered via async global->LDS copies; V manual (needs transpose).
// logits = QK^T/8 (WMMA), masked fp32 softmax, ctx = P*V (WMMA),
// scatter by perm (== inv_perm gather).
// ---------------------------------------------------------------------------
__global__ __launch_bounds__(128) void lsh_attn_kernel(
    const _Float16* __restrict__ qh, const _Float16* __restrict__ kh,
    const _Float16* __restrict__ vh, const unsigned char* __restrict__ mask,
    const int* __restrict__ perm, _Float16* __restrict__ ch) {
  const int blk = blockIdx.x, h = blockIdx.y, b = blockIdx.z;
  const int tid = threadIdx.x;
  __shared__ __align__(16) _Float16 qs[64 * PAD_];
  __shared__ __align__(16) _Float16 ks[64 * PAD_];
  __shared__ __align__(16) _Float16 vts[64 * PAD_];  // transposed: [dh][key]
  __shared__ __align__(16) _Float16 ps[64 * PAD_];   // softmax probs (f16)
  __shared__ float ls[64 * 68];                      // logits, padded rows
  __shared__ int   gidx[64];
  __shared__ float mk[64];

  if (tid < 64) {
    int g = perm[(size_t)b * N_ + blk * 64 + tid];
    gidx[tid] = g;
    mk[tid] = mask[(size_t)b * N_ + g] ? 1.0f : 0.0f;
  }
  __syncthreads();
  {
    int r = tid >> 1, seg = tid & 1;   // each thread: 32 halves of one row
    size_t base = ((size_t)(b * N_ + gidx[r])) * C_ + h * DH_ + seg * 32;
#pragma unroll
    for (int i = 0; i < 4; ++i) {      // async Q/K gather, 16B per lane
      async_copy_b128(lds_off(&qs[r * PAD_ + seg * 32 + i * 8]),
                      (unsigned long long)(uintptr_t)(qh + base + i * 8));
      async_copy_b128(lds_off(&ks[r * PAD_ + seg * 32 + i * 8]),
                      (unsigned long long)(uintptr_t)(kh + base + i * 8));
    }
    _Float16 tmp[32];                  // V: load then transpose into LDS
    uint4* tp = (uint4*)tmp;
    const uint4* sv = (const uint4*)(vh + base);
#pragma unroll
    for (int i = 0; i < 4; ++i) tp[i] = sv[i];
#pragma unroll
    for (int j = 0; j < 32; ++j) vts[(seg * 32 + j) * PAD_ + r] = tmp[j];
  }
  wait_async0();
  __syncthreads();

  const int lane = tid & 31, wave = tid >> 5;
  const int lm = lane & 15, lh2 = lane >> 4;

  // ---- logits = Q * K^T / sqrt(Dh), masked ----
  {
    v8f acc[4] = {};
#pragma unroll
    for (int k0 = 0; k0 < DH_; k0 += 32) {
      v16h a = load_a_frag(&qs[(wave * 16 + lm) * PAD_ + k0], lh2);
#pragma unroll
      for (int t = 0; t < 4; ++t) {
        v16h bf = load_b_frag(&ks[(t * 16 + lm) * PAD_ + k0], lh2);
        acc[t] = __builtin_amdgcn_wmma_f32_16x16x32_f16(false, a, false, bf,
                                                        (short)0, acc[t], false, false);
      }
    }
#pragma unroll
    for (int t = 0; t < 4; ++t) {
      int col = t * 16 + lm;
      float flag = mk[col];
#pragma unroll
      for (int e = 0; e < 8; ++e) {
        int row = wave * 16 + (lh2 ? e + 8 : e);
        float v = acc[t][e] * 0.125f;   // 1/sqrt(64)
        ls[row * 68 + col] = (flag != 0.0f) ? v : LNEG;
      }
    }
  }
  __syncthreads();

  // ---- fp32 softmax per row -> f16 probs ----
  if (tid < 64) {
    float* lr = ls + tid * 68;
    float m = lr[0];
    for (int c = 1; c < 64; ++c) m = fmaxf(m, lr[c]);
    float s = 0.f;
    for (int c = 0; c < 64; ++c) { float e = expf(lr[c] - m); s += e; lr[c] = e; }
    float inv = 1.0f / s;
    for (int c = 0; c < 64; ++c) ps[tid * PAD_ + c] = (_Float16)(lr[c] * inv);
  }
  __syncthreads();

  // ---- ctx = P * V, scatter back by perm ----
  {
    v8f acc[4] = {};
#pragma unroll
    for (int k0 = 0; k0 < 64; k0 += 32) {
      v16h a = load_a_frag(&ps[(wave * 16 + lm) * PAD_ + k0], lh2);
#pragma unroll
      for (int t = 0; t < 4; ++t) {
        v16h bf = load_b_frag(&vts[(t * 16 + lm) * PAD_ + k0], lh2);
        acc[t] = __builtin_amdgcn_wmma_f32_16x16x32_f16(false, a, false, bf,
                                                        (short)0, acc[t], false, false);
      }
    }
#pragma unroll
    for (int t = 0; t < 4; ++t) {
      int dh = t * 16 + lm;
#pragma unroll
      for (int e = 0; e < 8; ++e) {
        int row = wave * 16 + (lh2 ? e + 8 : e);
        int g = gidx[row];
        ch[((size_t)(b * N_ + g)) * C_ + h * DH_ + dh] = (_Float16)acc[t][e];
      }
    }
  }
}

// ---------------------------------------------------------------------------
extern "C" void kernel_launch(void* const* d_in, const int* in_sizes, int n_in,
                              void* d_out, int out_size, void* d_ws, size_t ws_size,
                              hipStream_t stream) {
  (void)in_sizes; (void)n_in; (void)out_size; (void)ws_size;
  const float* x          = (const float*)d_in[0];
  const unsigned char* mk = (const unsigned char*)d_in[1];  // jnp.bool_ = 1 byte
  const float* Wq = (const float*)d_in[2];
  const float* bq = (const float*)d_in[3];
  const float* Wk = (const float*)d_in[4];
  const float* bk = (const float*)d_in[5];
  const float* Wv = (const float*)d_in[6];
  const float* bv = (const float*)d_in[7];
  const float* Wo = (const float*)d_in[8];
  const float* bo = (const float*)d_in[9];
  const float* hp = (const float*)d_in[10];

  char* ws = (char*)d_ws;
  size_t ofs = 0;
  auto alloc = [&](size_t bytes) -> void* {
    void* p = ws + ofs;
    ofs = (ofs + bytes + 255) & ~(size_t)255;
    return p;
  };
  const size_t tok_elems = (size_t)BN_ * C_;
  _Float16* xh  = (_Float16*)alloc(tok_elems * 2);
  _Float16* qh  = (_Float16*)alloc(tok_elems * 2);
  _Float16* khb = (_Float16*)alloc(tok_elems * 2);
  _Float16* vhb = (_Float16*)alloc(tok_elems * 2);
  _Float16* ch  = (_Float16*)alloc(tok_elems * 2);
  _Float16* wqt = (_Float16*)alloc((size_t)C_ * C_ * 2);
  _Float16* wkt = (_Float16*)alloc((size_t)C_ * C_ * 2);
  _Float16* wvt = (_Float16*)alloc((size_t)C_ * C_ * 2);
  _Float16* wot = (_Float16*)alloc((size_t)C_ * C_ * 2);
  float* mu  = (float*)alloc((size_t)BN_ * 4);
  float* rs  = (float*)alloc((size_t)BN_ * 4);
  int*   bid = (int*)alloc((size_t)BN_ * 4);
  int*   prm = (int*)alloc((size_t)BN_ * 4);

  // 1) casts / transposed weights
  cast_x_kernel<<<(int)((tok_elems + 255) / 256), 256, 0, stream>>>(x, xh, (int)tok_elems);
  dim3 tb(32, 8), tg(C_ / 32, C_ / 32);
  transpose_cast_kernel<<<tg, tb, 0, stream>>>(Wq, wqt);
  transpose_cast_kernel<<<tg, tb, 0, stream>>>(Wk, wkt);
  transpose_cast_kernel<<<tg, tb, 0, stream>>>(Wv, wvt);
  transpose_cast_kernel<<<tg, tb, 0, stream>>>(Wo, wot);

  // 2) hashing + stable bucket sort (fp32 exact argmax semantics)
  ln_stats_kernel<<<BN_, 256, 0, stream>>>(x, mu, rs);
  lsh_bucket_kernel<<<BN_, NB_, 0, stream>>>(x, hp, mu, rs, bid);
  bucket_sort_kernel<<<B_, NB_, 0, stream>>>(bid, prm);

  // 3) Q/K/V projections (async-LDS double-buffered WMMA GEMM, f16 out)
  dim3 gg(C_ / 64, BN_ / 128);
  gemm_wmma_kernel<true><<<gg, 256, 0, stream>>>(xh, wqt, bq, qh, BN_, C_, C_);
  gemm_wmma_kernel<true><<<gg, 256, 0, stream>>>(xh, wkt, bk, khb, BN_, C_, C_);
  gemm_wmma_kernel<true><<<gg, 256, 0, stream>>>(xh, wvt, bv, vhb, BN_, C_, C_);

  // 4) block-local attention (gather/scatter by perm)
  lsh_attn_kernel<<<dim3(N_ / 64, H_, B_), 128, 0, stream>>>(qh, khb, vhb, mk, prm, ch);

  // 5) output projection (WMMA, fp32 out -> d_out)
  gemm_wmma_kernel<false><<<gg, 256, 0, stream>>>(ch, wot, bo, d_out, BN_, C_, C_);
}
